// RecformerForSeqRec_35725537968812
// MI455X (gfx1250) — matrix operand order repaired and checked by reading, not compiled, not run on hardware
//
#include <hip/hip_runtime.h>

#define B_SESS   16
#define S_ITEMS  20
#define SPAD     32          // pad session dim to 2 WMMA N-tiles
#define L_TOK    64
#define H_DIM    768
#define T_TOK    (S_ITEMS * L_TOK)   // 1280
#define NCAT     50000
#define NT_TILES (NCAT / 16)         // 3125
#define INV_TEMP 20.0f
#define EPS_F    1e-8f
#define LDSTRIDE 776                 // 768 + 8 halfs pad -> conflict-light LDS rows

typedef _Float16 v16h __attribute__((ext_vector_type(16)));
typedef _Float16 v8h  __attribute__((ext_vector_type(8)));
typedef float    v8f  __attribute__((ext_vector_type(8)));

__device__ __forceinline__ float block_reduce_sum(float v, float* sm) {
  int tid = threadIdx.x;
  sm[tid] = v;
  __syncthreads();
  for (int off = 128; off > 0; off >>= 1) {
    if (tid < off) sm[tid] += sm[tid + off];
    __syncthreads();
  }
  float r = sm[0];
  __syncthreads();
  return r;
}

// One block per (s, b): nan-mean pool 64 tokens, L2-normalize, emit f16 row.
// Rows s in [20,32) and invalid items are written as zeros (valid flag = 0).
__global__ void pool_norm_kernel(const float* __restrict__ hidden,
                                 const int* __restrict__ pos,
                                 _Float16* __restrict__ pooledN,
                                 int* __restrict__ valid) {
  __shared__ float sm[256];
  const int s   = blockIdx.x;   // 0..31
  const int b   = blockIdx.y;
  const int tid = threadIdx.x;
  _Float16* outRow = pooledN + ((size_t)b * SPAD + s) * H_DIM;

  bool isValid = false;
  if (s < S_ITEMS) isValid = (pos[b * T_TOK + s * L_TOK] == (s + 1));

  if (!isValid) {                       // uniform per block
    #pragma unroll
    for (int j = 0; j < 3; ++j) outRow[tid + j * 256] = (_Float16)0.0f;
    if (tid == 0) valid[b * SPAD + s] = 0;
    return;
  }

  const float* basep = hidden + ((size_t)b * T_TOK + (size_t)s * L_TOK) * H_DIM;
  float m[3];
  #pragma unroll
  for (int j = 0; j < 3; ++j) {
    const int h = tid + j * 256;
    float sum = 0.0f;
    for (int t = 0; t < L_TOK; ++t) sum += basep[(size_t)t * H_DIM + h];
    m[j] = sum * (1.0f / (float)L_TOK);
  }
  float ss = m[0] * m[0] + m[1] * m[1] + m[2] * m[2];
  ss = block_reduce_sum(ss, sm);
  const float inv = 1.0f / fmaxf(sqrtf(ss), EPS_F);
  #pragma unroll
  for (int j = 0; j < 3; ++j) outRow[tid + j * 256] = (_Float16)(m[j] * inv);
  if (tid == 0) valid[b * SPAD + s] = 1;
}

// One block per catalog row: L2-normalize, emit f16 row.
__global__ void emb_norm_kernel(const float* __restrict__ emb,
                                _Float16* __restrict__ embN) {
  __shared__ float sm[256];
  const int row = blockIdx.x;
  const int tid = threadIdx.x;
  const float* src = emb + (size_t)row * H_DIM;
  _Float16*    dst = embN + (size_t)row * H_DIM;
  float v[3];
  #pragma unroll
  for (int j = 0; j < 3; ++j) v[j] = src[tid + j * 256];
  float ss = v[0] * v[0] + v[1] * v[1] + v[2] * v[2];
  ss = block_reduce_sum(ss, sm);
  const float inv = 1.0f / fmaxf(sqrtf(ss), EPS_F);
  #pragma unroll
  for (int j = 0; j < 3; ++j) dst[tid + j * 256] = (_Float16)(v[j] * inv);
}

// A-fragment in contiguous k order (matches B): lane row = lane%16;
// elems 0-15 -> K = 16*(lane/16) + {0..15}.
__device__ __forceinline__ v16h load_afrag(const _Float16* p) {
  const v8h lo = *(const v8h*)p;
  const v8h hi = *(const v8h*)(p + 8);
  return __builtin_shufflevector(lo, hi, 0, 1, 2, 3, 4, 5, 6, 7,
                                 8, 9, 10, 11, 12, 13, 14, 15);
}

// Each wave: MU=2 catalog tiles (32 rows) x BU=2 sessions x 2 s-tiles
// -> 8 f32 16x16 accumulators. Pooled operand staged in padded LDS once
// per block; catalog operand streamed from global/L2 with WGP-scope
// prefetch.
__global__ void maxsim_wmma_kernel(const _Float16* __restrict__ embN,
                                   const _Float16* __restrict__ pooledN,
                                   const int* __restrict__ valid,
                                   float* __restrict__ out) {
  __shared__ _Float16 ldsB[2 * SPAD * LDSTRIDE];   // ~97 KB of 320 KB/WGP
  const int tid = threadIdx.x;
  const int b0  = blockIdx.y * 2;

  // Cooperative fill: pooled rows for sessions b0, b0+1 (v8h chunks).
  #pragma unroll 4
  for (int c = tid; c < 2 * SPAD * (H_DIM / 8); c += 256) {
    const int bb  = c / (SPAD * (H_DIM / 8));
    const int rem = c - bb * (SPAD * (H_DIM / 8));
    const int r   = rem / (H_DIM / 8);
    const int kc  = (rem - r * (H_DIM / 8)) * 8;
    const v8h v = *(const v8h*)(pooledN + ((size_t)(b0 + bb) * SPAD + r) * H_DIM + kc);
    *(v8h*)(ldsB + (size_t)(bb * SPAD + r) * LDSTRIDE + kc) = v;
  }
  __syncthreads();

  const int lane = tid & 31;
  const int wave = tid >> 5;
  const int half = lane >> 4;          // 0: lanes 0-15, 1: lanes 16-31
  const int l16  = lane & 15;
  const int t0   = (blockIdx.x * 8 + wave) * 2;   // first of two n-tiles

  const _Float16* a0Base = embN + (size_t)(t0 * 16 + l16) * H_DIM + (half ? 16 : 0);
  const _Float16* a1Base = a0Base + (size_t)16 * H_DIM;
  // B fragment (Kx16): column = lane%16; lanes 0-15 hold K{0..15}, 16-31 K{16..31}.
  const _Float16* lb = ldsB + (size_t)l16 * LDSTRIDE + (half ? 16 : 0);

  v8f acc[2][2][2];                    // [mu][bb][s-tile]
  #pragma unroll
  for (int mu = 0; mu < 2; ++mu)
    #pragma unroll
    for (int bb = 0; bb < 2; ++bb)
      #pragma unroll
      for (int st = 0; st < 2; ++st)
        #pragma unroll
        for (int e = 0; e < 8; ++e) acc[mu][bb][st][e] = 0.0f;

  for (int kb = 0; kb < H_DIM; kb += 32) {
    const v16h a0 = load_afrag(a0Base + kb);
    const v16h a1 = load_afrag(a1Base + kb);
    __builtin_prefetch(a0Base + kb + 64, 0, 3);   // WGP-scope: fill WGP$ + L2
    __builtin_prefetch(a1Base + kb + 64, 0, 3);
    #pragma unroll
    for (int bb = 0; bb < 2; ++bb) {
      const _Float16* p = lb + (size_t)bb * SPAD * LDSTRIDE + kb;
      const v16h bf0 = *(const v16h*)(p);
      const v16h bf1 = *(const v16h*)(p + 16 * LDSTRIDE);
      acc[0][bb][0] = __builtin_amdgcn_wmma_f32_16x16x32_f16(
          false, a0, false, bf0, (short)0, acc[0][bb][0], false, false);
      acc[0][bb][1] = __builtin_amdgcn_wmma_f32_16x16x32_f16(
          false, a0, false, bf1, (short)0, acc[0][bb][1], false, false);
      acc[1][bb][0] = __builtin_amdgcn_wmma_f32_16x16x32_f16(
          false, a1, false, bf0, (short)0, acc[1][bb][0], false, false);
      acc[1][bb][1] = __builtin_amdgcn_wmma_f32_16x16x32_f16(
          false, a1, false, bf1, (short)0, acc[1][bb][1], false, false);
    }
  }

  // D layout: lane column = lane%16; VGPR r holds row r + 8*(lane/16).
  const float NEG_INF = -__builtin_inff();
  #pragma unroll
  for (int bb = 0; bb < 2; ++bb) {
    const int* vf = valid + (b0 + bb) * SPAD;
    const bool v0ok = vf[l16] != 0;
    const bool v1ok = vf[l16 + 16] != 0;   // s >= 20 always invalid
    #pragma unroll
    for (int mu = 0; mu < 2; ++mu) {
      const int tile = t0 + mu;
      #pragma unroll
      for (int r = 0; r < 8; ++r) {
        float m = fmaxf(v0ok ? acc[mu][bb][0][r] : NEG_INF,
                        v1ok ? acc[mu][bb][1][r] : NEG_INF);
        m = fmaxf(m, __shfl_xor(m, 8, 32));   // reduce each 16-lane group
        m = fmaxf(m, __shfl_xor(m, 4, 32));
        m = fmaxf(m, __shfl_xor(m, 2, 32));
        m = fmaxf(m, __shfl_xor(m, 1, 32));
        if (l16 == 0 && tile < NT_TILES) {
          out[(size_t)(b0 + bb) * NCAT + tile * 16 + r + half * 8] = m * INV_TEMP;
        }
      }
    }
  }
}

extern "C" void kernel_launch(void* const* d_in, const int* in_sizes, int n_in,
                              void* d_out, int out_size, void* d_ws, size_t ws_size,
                              hipStream_t stream) {
  (void)in_sizes; (void)n_in; (void)out_size; (void)ws_size;
  const float* hidden = (const float*)d_in[0];   // [B,T,H] f32
  const float* emb    = (const float*)d_in[1];   // [N,H]   f32
  const int*   pos    = (const int*)d_in[3];     // [B,T]   i32

  char* ws = (char*)d_ws;
  const size_t embBytes    = (size_t)NCAT * H_DIM * sizeof(_Float16);      // 76.8 MB
  const size_t pooledBytes = (size_t)B_SESS * SPAD * H_DIM * sizeof(_Float16);
  _Float16* embN    = (_Float16*)ws;
  _Float16* pooledN = (_Float16*)(ws + embBytes);
  int*      valid   = (int*)(ws + embBytes + pooledBytes);

  pool_norm_kernel<<<dim3(SPAD, B_SESS), 256, 0, stream>>>(hidden, pos, pooledN, valid);
  emb_norm_kernel<<<dim3(NCAT), 256, 0, stream>>>(emb, embN);

  // 8 waves/block, each wave: 2 catalog tiles x 2 sessions; grid covers
  // ceil(3125/16)=196 tile-groups x 8 session-pairs.
  dim3 gridC((NT_TILES + 15) / 16, B_SESS / 2);
  maxsim_wmma_kernel<<<gridC, 256, 0, stream>>>(embN, pooledN, valid, (float*)d_out);
}
